// MGAN_41970420416830
// MI455X (gfx1250) — compile-verified, matrix operand order checked by reading, not run on hardware
//
#include <hip/hip_runtime.h>
#include <hip/hip_bf16.h>
#include <stdint.h>
#include <math.h>

// Problem constants (match reference)
#define B_   64
#define LC_  256
#define LA_  32
#define E_   300
#define KP_  320     // E padded to multiple of 32 for WMMA K
#define H_   512
#define H2_  1024    // 2H
#define H4_  2048    // 4H
#define P_   3

typedef __bf16 bf16_t;
typedef __attribute__((ext_vector_type(16))) __bf16 bf16x16;
typedef __attribute__((ext_vector_type(8)))  float  f32x8;
typedef __attribute__((ext_vector_type(4)))  uint32_t u32x4;
typedef __attribute__((ext_vector_type(8)))  int      i32x8;
typedef __attribute__((ext_vector_type(4)))  int      i32x4;

// ---------------------------------------------------------------------------
// WMMA helpers (wave32, V_WMMA_F32_16X16X32_BF16)
// A fragment: 16x32 (MxK).  lane L: m=L&15, kgroup=L>>4,
//   halves 0..7  -> k = kgroup*8 + h
//   halves 8..15 -> k = 16 + kgroup*8 + (h-8)
// B fragment: 32x16 (KxN), same pattern with n=L&15 (row of W holds K).
// C/D: lane L holds column n=L&15; rows m = (L>=16?8:0)+i for vgpr i.
// Works for both global and LDS source pointers (addrspace inferred after
// inlining; LDS sources lower to ds_load).
// ---------------------------------------------------------------------------
__device__ __forceinline__ bf16x16 ld_frag(const bf16_t* p, int ld, int row0, int k0) {
  int lane = threadIdx.x & 31;
  const bf16_t* base = p + (size_t)(row0 + (lane & 15)) * ld + k0 + ((lane >> 4) & 1) * 8;
  bf16x16 f;
#pragma unroll
  for (int h = 0; h < 8; ++h) { f[h] = base[h]; f[h + 8] = base[16 + h]; }
  return f;
}

__device__ __forceinline__ f32x8 wmma_bf(bf16x16 a, bf16x16 b, f32x8 c) {
  return __builtin_amdgcn_wmma_f32_16x16x32_bf16(false, a, false, b, (short)0, c, false, false);
}

__device__ __forceinline__ float sigmoidf_(float x) { return 1.0f / (1.0f + expf(-x)); }

// ---------------------------------------------------------------------------
// K: sequence lengths (count of nonzero indices per row)
// lens[0..63]=ctx_len, [64..127]=asp_len, [128..191]=left_len
// ---------------------------------------------------------------------------
__global__ __launch_bounds__(256) void k_lengths(const int* traw, const int* aidx,
                                                 const int* tleft, int* lens) {
  __shared__ int red[256];
  int b = blockIdx.x, t = threadIdx.x;
  int v = (traw[b * LC_ + t] != 0) ? 1 : 0;
  red[t] = v; __syncthreads();
  for (int s = 128; s > 0; s >>= 1) { if (t < s) red[t] += red[t + s]; __syncthreads(); }
  if (t == 0) lens[b] = red[0];
  __syncthreads();
  v = (t < LA_ && aidx[b * LA_ + t] != 0) ? 1 : 0;
  red[t] = v; __syncthreads();
  for (int s = 128; s > 0; s >>= 1) { if (t < s) red[t] += red[t + s]; __syncthreads(); }
  if (t == 0) lens[B_ + b] = red[0];
  __syncthreads();
  v = (tleft[b * LC_ + t] != 0) ? 1 : 0;
  red[t] = v; __syncthreads();
  for (int s = 128; s > 0; s >>= 1) { if (t < s) red[t] += red[t + s]; __syncthreads(); }
  if (t == 0) lens[2 * B_ + b] = red[0];
}

// ---------------------------------------------------------------------------
// K: embedding gather -> padded bf16 activations  X[r, 0..319]
// ---------------------------------------------------------------------------
__global__ __launch_bounds__(64) void k_embed(const int* traw, const int* aidx,
                                              const float* emb, bf16_t* Xc, bf16_t* Xa) {
  int r = blockIdx.x;
  const int* src; bf16_t* dst; int n;
  if (r < B_ * LC_) { n = r; src = traw; dst = Xc + (size_t)n * KP_; }
  else              { n = r - B_ * LC_; src = aidx; dst = Xa + (size_t)n * KP_; }
  int idx = src[n];
  const float* er = emb + (size_t)idx * E_;
  for (int k = threadIdx.x; k < KP_; k += blockDim.x)
    dst[k] = (k < E_) ? (bf16_t)er[k] : (bf16_t)0.0f;
}

// ---------------------------------------------------------------------------
// K: f32 -> bf16 weight convert with K padding
// ---------------------------------------------------------------------------
__global__ __launch_bounds__(64) void k_cvt_pad(const float* src, bf16_t* dst,
                                                int cols, int colsp) {
  int r = blockIdx.x;
  for (int k = threadIdx.x; k < colsp; k += blockDim.x)
    dst[(size_t)r * colsp + k] = (k < cols) ? (bf16_t)src[(size_t)r * cols + k] : (bf16_t)0.0f;
}

__global__ void k_zero32(uint32_t* p, int n) {
  int i = blockIdx.x * blockDim.x + threadIdx.x;
  if (i < n) p[i] = 0;
}

// ---------------------------------------------------------------------------
// K: input projection GEMM.  G[M,4H] = X[M,KP] * W[4H,KP]^T + (bih+bhh)
// grid = (4H/16, M/128); block = 256 (8 waves, each a 16x16 tile)
// ---------------------------------------------------------------------------
__global__ __launch_bounds__(256) void k_gemm_inproj(const bf16_t* __restrict__ X,
                                                     const bf16_t* __restrict__ W,
                                                     const float* __restrict__ bih,
                                                     const float* __restrict__ bhh,
                                                     bf16_t* __restrict__ G) {
  int col0 = blockIdx.x * 16;
  int wave = threadIdx.x >> 5;
  int row0 = blockIdx.y * 128 + wave * 16;
  int lane = threadIdx.x & 31;
  f32x8 acc = {0.f, 0.f, 0.f, 0.f, 0.f, 0.f, 0.f, 0.f};
#pragma unroll 2
  for (int k = 0; k < KP_; k += 32) {
    if (k + 32 < KP_) {
      // speculative prefetch of the next K tile (global_prefetch_b8)
      __builtin_prefetch(X + (size_t)(row0 + (lane & 15)) * KP_ + k + 32, 0, 1);
      __builtin_prefetch(W + (size_t)(col0 + (lane & 15)) * KP_ + k + 32, 0, 1);
    }
    bf16x16 a = ld_frag(X, KP_, row0, k);
    bf16x16 b = ld_frag(W, KP_, col0, k);
    acc = wmma_bf(a, b, acc);
  }
  int n = col0 + (lane & 15);
  int mb = row0 + ((lane >> 4) & 1) * 8;
  float bias = bih[n] + bhh[n];
#pragma unroll
  for (int i = 0; i < 8; ++i)
    G[(size_t)(mb + i) * H4_ + n] = (bf16_t)(acc[i] + bias);
}

// ---------------------------------------------------------------------------
// K: one LSTM timestep for both directions of one BiLSTM.
// grid = (H/32, 1, 2); block = 256.
// Block (j0, dir) computes gate columns {g*512 + j0 .. j0+31 : g=0..3} as a
// 64x128 WMMA GEMM over K=512 (h state), adds precomputed input pre-gates G,
// stages into LDS, then performs the cell update with packed-seq masking,
// backward un-reversal, and fused location weighting on the output write.
// The shared h state [64,512] is staged through LDS in 16 KB phases by the
// Tensor Data Mover (tensor_load_to_lds, one DMA per block issued by wave 0,
// completion on TENSORcnt) so WMMA A-fragments come from ds_load and the
// VALU/VGPRs stay free during the copy.
// h ping-pongs between launches (no cross-block race inside a launch).
// ---------------------------------------------------------------------------
#define KPH_ 128   // K-phase width staged in LDS (64 x 128 bf16 = 16 KB)

struct LstmArgs {
  const bf16_t* G0;  const bf16_t* G1;   // pre-gates [B*T, 4H] per dir
  const bf16_t* W0;  const bf16_t* W1;   // whh bf16 [4H, H] per dir
  const bf16_t* hin0; const bf16_t* hin1;
  bf16_t* hout0;      bf16_t* hout1;
  float*  c0;         float*  c1;
  bf16_t* out;                            // [B, T, 2H] (fwd cols 0:H, bwd H:2H)
  const int* lens; const int* leftlen; const int* asplen;
  int T; int t; int use_locw;
};

__global__ __launch_bounds__(256) void k_lstm_step(LstmArgs A) {
  __shared__ float gates[64 * 128];                       // 32 KB
  __shared__ __align__(16) bf16_t hs[64 * KPH_];          // 16 KB h stage
  int dir = blockIdx.z;
  int j0 = blockIdx.x * 32;
  const bf16_t* G   = dir ? A.G1 : A.G0;
  const bf16_t* Wh  = dir ? A.W1 : A.W0;
  const bf16_t* hin = dir ? A.hin1 : A.hin0;
  bf16_t* hout      = dir ? A.hout1 : A.hout0;
  float*  cst       = dir ? A.c1 : A.c0;

  int wave = threadIdx.x >> 5, lane = threadIdx.x & 31;
  // wave w owns block-local cols [w*16, w*16+16); whh row base:
  int row0 = (wave >> 1) * H_ + j0 + (wave & 1) * 16;
  f32x8 acc0 = {0.f,0.f,0.f,0.f,0.f,0.f,0.f,0.f};
  f32x8 acc1 = acc0, acc2 = acc0, acc3 = acc0;

  for (int kh = 0; kh < H_; kh += KPH_) {
    __syncthreads();   // protect hs reuse across phases
    if (threadIdx.x < 32) {
      // TDM: DMA the 64x128 bf16 tile h[:, kh:kh+128] of the 64x512 tensor
      // into LDS 'hs', packed row-major (tile rows contiguous, stride 128).
      uint64_t ga = (uint64_t)(uintptr_t)(hin + kh);        // tile start
      uint32_t lds = (uint32_t)(uintptr_t)(&hs[0]);         // LDS byte offset
      u32x4 g0;
      g0[0] = 1u;                                           // count=1, user D#
      g0[1] = lds;                                          // lds_addr
      g0[2] = (uint32_t)ga;                                 // global_addr[31:0]
      g0[3] = (uint32_t)((ga >> 32) & 0x01FFFFFFu) | (2u << 30); // [56:32]|type=2
      i32x8 g1;
      g1[0] = 0x00010000;            // wg_mask=0, data_size=2B, no iter/pad
      g1[1] = 0x02000000;            // tensor_dim0=512 (low16 at bits 63:48)
      g1[2] = 0x00400000;            // tensor_dim0 hi=0; tensor_dim1=64
      g1[3] = (int)(128u << 16);     // tensor_dim1 hi=0; tile_dim0=128
      g1[4] = 64;                    // tile_dim1=64; tile_dim2=0
      g1[5] = 512;                   // tensor_dim0_stride=512 (elements)
      g1[6] = 0; g1[7] = 0;          // dim1_stride unused (2D)
      i32x4 gz = {0, 0, 0, 0};
      i32x8 gz8 = {0, 0, 0, 0, 0, 0, 0, 0};
      __builtin_amdgcn_tensor_load_to_lds(g0, g1, gz, gz, gz8, 0);
      __builtin_amdgcn_s_wait_tensorcnt(0);
    }
    __syncthreads();
#pragma unroll
    for (int k = 0; k < KPH_; k += 32) {
      if (kh + k + 32 < H_)
        __builtin_prefetch(Wh + (size_t)(row0 + (lane & 15)) * H_ + kh + k + 32, 0, 1);
      bf16x16 bfrag = ld_frag(Wh, H_, row0, kh + k);
      acc0 = wmma_bf(ld_frag(hs, KPH_,  0, k), bfrag, acc0);
      acc1 = wmma_bf(ld_frag(hs, KPH_, 16, k), bfrag, acc1);
      acc2 = wmma_bf(ld_frag(hs, KPH_, 32, k), bfrag, acc2);
      acc3 = wmma_bf(ld_frag(hs, KPH_, 48, k), bfrag, acc3);
    }
  }

  // Epilogue: add input pre-gates and stage into LDS
  int n = lane & 15;
  int c_local = wave * 16 + n;                       // 0..127
  int gcol = (c_local >> 5) * H_ + j0 + (c_local & 31);
  int hi16 = ((lane >> 4) & 1) * 8;
  const f32x8* accs[4] = {&acc0, &acc1, &acc2, &acc3};
#pragma unroll
  for (int mt = 0; mt < 4; ++mt) {
    int mb = mt * 16 + hi16;
#pragma unroll
    for (int i = 0; i < 8; ++i) {
      int b = mb + i;
      int len = A.lens[b];
      int tpos = A.t;
      if (dir == 1) tpos = (A.t < len) ? (len - 1 - A.t) : A.t;
      float gin = (float)G[(size_t)(b * A.T + tpos) * H4_ + gcol];
      gates[b * 128 + c_local] = (*accs[mt])[i] + gin;
    }
  }
  __syncthreads();

  // Cell update for 64 batch x 32 hidden units
  for (int e = threadIdx.x; e < 64 * 32; e += blockDim.x) {
    int b = e >> 5, j = e & 31, jH = j0 + j;
    float iv = sigmoidf_(gates[b * 128 + j]);
    float fv = sigmoidf_(gates[b * 128 + 32 + j]);
    float gv = tanhf(gates[b * 128 + 64 + j]);
    float ov = sigmoidf_(gates[b * 128 + 96 + j]);
    float cold = cst[b * H_ + jH];
    float cn = fv * cold + iv * gv;
    float hn = ov * tanhf(cn);
    int len = A.lens[b];
    bool valid = (A.t < len);
    cst[b * H_ + jH] = valid ? cn : cold;
    float hprev = (float)hin[b * H_ + jH];
    hout[b * H_ + jH] = (bf16_t)(valid ? hn : hprev);
    int pos = valid ? ((dir == 0) ? A.t : (len - 1 - A.t)) : A.t;
    float oval = valid ? hn : 0.0f;
    if (A.use_locw) {
      float s = (float)A.leftlen[b];
      float al = (float)A.asplen[b];
      float ee = s + al - 1.0f;
      float sent = (float)A.T - al;
      float jf = (float)pos;
      float w = (jf < s) ? (1.0f - (s - jf) / sent)
                         : ((jf <= ee) ? 0.0f : (1.0f - (jf - ee) / sent));
      oval *= w;
    }
    A.out[((size_t)b * A.T + pos) * H2_ + dir * H_ + jH] = (bf16_t)oval;
  }
}

// ---------------------------------------------------------------------------
// K: mean pool over time:  pool[b,d] = sum_t mat[b,t,d] / lens[b]
// ---------------------------------------------------------------------------
__global__ __launch_bounds__(256) void k_pool(const bf16_t* mat, const int* lens,
                                              float* pool, int L) {
  int b = blockIdx.x;
  float inv = 1.0f / (float)lens[b];
  for (int d = threadIdx.x; d < H2_; d += blockDim.x) {
    float s = 0.f;
    for (int t = 0; t < L; ++t) s += (float)mat[((size_t)b * L + t) * H2_ + d];
    pool[b * H2_ + d] = s * inv;
  }
}

// K: u[b,d] = sum_k Wm[d,k] * pool[b,k]   (grid = (H2/256, B))
__global__ __launch_bounds__(256) void k_matvec_u(const float* Wm, const float* pool,
                                                  float* u) {
  int b = blockIdx.y;
  int d = blockIdx.x * blockDim.x + threadIdx.x;
  const float* wr = Wm + (size_t)d * H2_;
  const float* pb = pool + b * H2_;
  float s = 0.f;
  for (int k = 0; k < H2_; ++k) s += wr[k] * pb[k];
  u[b * H2_ + d] = s;
}

// K: out[b,i] = dot(mat[b,i,:], w[b*wstride + :])
__global__ __launch_bounds__(256) void k_proj_w(const bf16_t* mat, const float* w,
                                                int wstride, float* out, int L) {
  int b = blockIdx.x;
  const float* wb = w + (size_t)b * wstride;
  for (int i = threadIdx.x; i < L; i += blockDim.x) {
    const bf16_t* r = mat + ((size_t)b * L + i) * H2_;
    float s = 0.f;
    for (int k = 0; k < H2_; ++k) s += (float)r[k] * wb[k];
    out[b * L + i] = s;
  }
}

// K: aspw3[b,j,d] = asp[b,j,d] * w3[d]
__global__ __launch_bounds__(256) void k_scale_w3(const bf16_t* asp, const float* w3,
                                                  bf16_t* out) {
  size_t i = (size_t)blockIdx.x * 256 + threadIdx.x;
  if (i < (size_t)B_ * LA_ * H2_) {
    int d = (int)(i & (H2_ - 1));
    out[i] = (bf16_t)((float)asp[i] * w3[d]);
  }
}

// ---------------------------------------------------------------------------
// K: alignment batched WMMA GEMM: align[b,i,j] = ctx[b,i,:].aspw3[b,j,:]
//    + cw1[b,i] + aw2[b,j].   grid = (LA/16, LC/128, B); block = 256
// ---------------------------------------------------------------------------
__global__ __launch_bounds__(256) void k_align_gemm(const bf16_t* ctx, const bf16_t* aspw3,
                                                    const float* cw1, const float* aw2,
                                                    float* align) {
  int b = blockIdx.z;
  int col0 = blockIdx.x * 16;
  int wave = threadIdx.x >> 5;
  int row0 = blockIdx.y * 128 + wave * 16;
  int lane = threadIdx.x & 31;
  const bf16_t* Am = ctx + (size_t)b * LC_ * H2_;
  const bf16_t* Bm = aspw3 + (size_t)b * LA_ * H2_;
  f32x8 acc = {0.f,0.f,0.f,0.f,0.f,0.f,0.f,0.f};
  for (int k = 0; k < H2_; k += 32) {
    if (k + 32 < H2_)
      __builtin_prefetch(Am + (size_t)(row0 + (lane & 15)) * H2_ + k + 32, 0, 1);
    acc = wmma_bf(ld_frag(Am, H2_, row0, k), ld_frag(Bm, H2_, col0, k), acc);
  }
  int n = col0 + (lane & 15);
  int mb = row0 + ((lane >> 4) & 1) * 8;
#pragma unroll
  for (int i = 0; i < 8; ++i) {
    int m = mb + i;
    align[((size_t)b * LC_ + m) * LA_ + n] = acc[i] + cw1[b * LC_ + m] + aw2[b * LA_ + n];
  }
}

__global__ __launch_bounds__(256) void k_rowmax(const float* align, float* out) {
  int b = blockIdx.x;
  for (int i = threadIdx.x; i < LC_; i += blockDim.x) {
    const float* r = align + ((size_t)b * LC_ + i) * LA_;
    float m = r[0];
    for (int j = 1; j < LA_; ++j) m = fmaxf(m, r[j]);
    out[b * LC_ + i] = m;
  }
}

__global__ __launch_bounds__(256) void k_colmax(const float* align, float* out) {
  int b = blockIdx.x;
  for (int j = threadIdx.x; j < LA_; j += blockDim.x) {
    float m = align[(size_t)b * LC_ * LA_ + j];
    for (int i = 1; i < LC_; ++i) m = fmaxf(m, align[((size_t)b * LC_ + i) * LA_ + j]);
    out[b * LA_ + j] = m;
  }
}

// K: out[b,d] = sum_i softmax(score[b,:])[i] * mat[b,i,d]   (L <= 256)
__global__ __launch_bounds__(256) void k_softmax_wsum(const float* score, const bf16_t* mat,
                                                      float* out, int L) {
  __shared__ float sc[256];
  __shared__ float red[256];
  int b = blockIdx.x, t = threadIdx.x;
  float v = (t < L) ? score[b * L + t] : -1e30f;
  red[t] = v; __syncthreads();
  for (int s = 128; s > 0; s >>= 1) { if (t < s) red[t] = fmaxf(red[t], red[t + s]); __syncthreads(); }
  float mx = red[0]; __syncthreads();
  float e = (t < L) ? expf(v - mx) : 0.f;
  sc[t] = e; red[t] = e; __syncthreads();
  for (int s = 128; s > 0; s >>= 1) { if (t < s) red[t] += red[t + s]; __syncthreads(); }
  float inv = 1.0f / red[0];
  __syncthreads();
  for (int d = t; d < H2_; d += blockDim.x) {
    float s = 0.f;
    for (int i = 0; i < L; ++i) s += sc[i] * (float)mat[((size_t)b * L + i) * H2_ + d];
    out[b * H2_ + d] = s * inv;
  }
}

// K: final dense: out[b,p] = [c_a2c|f_a2c|f_c2a|c_c2a] . dw[p,:] + db[p]
__global__ __launch_bounds__(256) void k_dense(const float* c_a2c, const float* f_a2c,
                                               const float* f_c2a, const float* c_c2a,
                                               const float* dw, const float* db, float* out) {
  __shared__ float red[256];
  int b = blockIdx.x, t = threadIdx.x;
  const float* segs[4] = {c_a2c + (size_t)b * H2_, f_a2c + (size_t)b * H2_,
                          f_c2a + (size_t)b * H2_, c_c2a + (size_t)b * H2_};
  for (int p = 0; p < P_; ++p) {
    float s = 0.f;
    for (int k = t; k < 4 * H2_; k += blockDim.x)
      s += segs[k >> 10][k & (H2_ - 1)] * dw[(size_t)p * 4 * H2_ + k];
    red[t] = s; __syncthreads();
    for (int st = 128; st > 0; st >>= 1) { if (t < st) red[t] += red[t + st]; __syncthreads(); }
    if (t == 0) out[b * P_ + p] = red[0] + db[p];
    __syncthreads();
  }
}

// ---------------------------------------------------------------------------
// Host orchestration
// ---------------------------------------------------------------------------
extern "C" void kernel_launch(void* const* d_in, const int* in_sizes, int n_in,
                              void* d_out, int out_size, void* d_ws, size_t ws_size,
                              hipStream_t stream) {
  const int*   traw   = (const int*)d_in[0];
  const int*   aidx   = (const int*)d_in[1];
  const int*   tleft  = (const int*)d_in[2];
  const float* emb    = (const float*)d_in[3];
  const float* w_a2c  = (const float*)d_in[4];
  const float* w_c2a  = (const float*)d_in[5];
  const float* w_u    = (const float*)d_in[6];
  const float* dw     = (const float*)d_in[7];
  const float* db     = (const float*)d_in[8];
  // instances: 0=ctx_f, 1=ctx_b, 2=asp_f, 3=asp_b
  const float *wih[4], *whh[4], *bih[4], *bhh[4];
  for (int i = 0; i < 4; ++i) {
    wih[i] = (const float*)d_in[9 + 4 * i];
    whh[i] = (const float*)d_in[10 + 4 * i];
    bih[i] = (const float*)d_in[11 + 4 * i];
    bhh[i] = (const float*)d_in[12 + 4 * i];
  }

  // ---- workspace carve (~225 MB) ----
  char* wp = (char*)d_ws;
  auto alloc = [&](size_t bytes) -> char* {
    char* r = wp; wp += (bytes + 255) & ~(size_t)255; return r;
  };
  int* lens = (int*)alloc(3 * B_ * sizeof(int));                   // ctx,asp,left
  bf16_t* Xc = (bf16_t*)alloc((size_t)B_ * LC_ * KP_ * 2);
  bf16_t* Xa = (bf16_t*)alloc((size_t)B_ * LA_ * KP_ * 2);
  bf16_t *wihb[4], *whhb[4];
  for (int i = 0; i < 4; ++i) {
    wihb[i] = (bf16_t*)alloc((size_t)H4_ * KP_ * 2);
    whhb[i] = (bf16_t*)alloc((size_t)H4_ * H_ * 2);
  }
  bf16_t* Gc[2]; bf16_t* Ga[2];
  Gc[0] = (bf16_t*)alloc((size_t)B_ * LC_ * H4_ * 2);
  Gc[1] = (bf16_t*)alloc((size_t)B_ * LC_ * H4_ * 2);
  Ga[0] = (bf16_t*)alloc((size_t)B_ * LA_ * H4_ * 2);
  Ga[1] = (bf16_t*)alloc((size_t)B_ * LA_ * H4_ * 2);
  bf16_t* ctx_out = (bf16_t*)alloc((size_t)B_ * LC_ * H2_ * 2);    // location-weighted
  bf16_t* asp_out = (bf16_t*)alloc((size_t)B_ * LA_ * H2_ * 2);
  bf16_t* hbuf = (bf16_t*)alloc((size_t)4 * 2 * B_ * H_ * 2);      // [inst][pp][B*H]
  float*  cbuf = (float*)alloc((size_t)4 * B_ * H_ * 4);
  float* ctx_pool = (float*)alloc((size_t)B_ * H2_ * 4);
  float* asp_pool = (float*)alloc((size_t)B_ * H2_ * 4);
  float* u_a2c = (float*)alloc((size_t)B_ * H2_ * 4);
  float* u_c2a = (float*)alloc((size_t)B_ * H2_ * 4);
  float* cw1 = (float*)alloc((size_t)B_ * LC_ * 4);
  float* aw2 = (float*)alloc((size_t)B_ * LA_ * 4);
  bf16_t* aspw3 = (bf16_t*)alloc((size_t)B_ * LA_ * H2_ * 2);
  float* align = (float*)alloc((size_t)B_ * LC_ * LA_ * 4);
  float* rmax = (float*)alloc((size_t)B_ * LC_ * 4);
  float* cmax = (float*)alloc((size_t)B_ * LA_ * 4);
  float* sc_a2c = (float*)alloc((size_t)B_ * LC_ * 4);
  float* sc_c2a = (float*)alloc((size_t)B_ * LA_ * 4);
  float* f_a2c = (float*)alloc((size_t)B_ * H2_ * 4);
  float* f_c2a = (float*)alloc((size_t)B_ * H2_ * 4);
  float* c_a2c = (float*)alloc((size_t)B_ * H2_ * 4);
  float* c_c2a = (float*)alloc((size_t)B_ * H2_ * 4);
  auto Hat = [&](int inst, int pp) { return hbuf + ((size_t)(inst * 2 + pp)) * B_ * H_; };
  auto Cat = [&](int inst) { return cbuf + (size_t)inst * B_ * H_; };

  // ---- prep ----
  k_lengths<<<B_, 256, 0, stream>>>(traw, aidx, tleft, lens);
  k_embed<<<B_ * LC_ + B_ * LA_, 64, 0, stream>>>(traw, aidx, emb, Xc, Xa);
  for (int i = 0; i < 4; ++i) {
    k_cvt_pad<<<H4_, 64, 0, stream>>>(wih[i], wihb[i], E_, KP_);
    k_cvt_pad<<<H4_, 64, 0, stream>>>(whh[i], whhb[i], H_, H_);
  }
  {
    int nh = (int)((size_t)4 * 2 * B_ * H_ * 2 / 4);
    int nc = (int)((size_t)4 * B_ * H_ * 4 / 4);
    k_zero32<<<(nh + 255) / 256, 256, 0, stream>>>((uint32_t*)hbuf, nh);
    k_zero32<<<(nc + 255) / 256, 256, 0, stream>>>((uint32_t*)cbuf, nc);
  }

  // ---- input projections (big WMMA GEMMs) ----
  k_gemm_inproj<<<dim3(H4_ / 16, (B_ * LC_) / 128), 256, 0, stream>>>(Xc, wihb[0], bih[0], bhh[0], Gc[0]);
  k_gemm_inproj<<<dim3(H4_ / 16, (B_ * LC_) / 128), 256, 0, stream>>>(Xc, wihb[1], bih[1], bhh[1], Gc[1]);
  k_gemm_inproj<<<dim3(H4_ / 16, (B_ * LA_) / 128), 256, 0, stream>>>(Xa, wihb[2], bih[2], bhh[2], Ga[0]);
  k_gemm_inproj<<<dim3(H4_ / 16, (B_ * LA_) / 128), 256, 0, stream>>>(Xa, wihb[3], bih[3], bhh[3], Ga[1]);

  // ---- context BiLSTM recurrence (256 steps, both dirs per launch) ----
  for (int t = 0; t < LC_; ++t) {
    LstmArgs a;
    a.G0 = Gc[0]; a.G1 = Gc[1]; a.W0 = whhb[0]; a.W1 = whhb[1];
    a.hin0 = Hat(0, t & 1);  a.hin1 = Hat(1, t & 1);
    a.hout0 = Hat(0, (t + 1) & 1); a.hout1 = Hat(1, (t + 1) & 1);
    a.c0 = Cat(0); a.c1 = Cat(1);
    a.out = ctx_out; a.lens = lens; a.leftlen = lens + 2 * B_; a.asplen = lens + B_;
    a.T = LC_; a.t = t; a.use_locw = 1;
    k_lstm_step<<<dim3(H_ / 32, 1, 2), 256, 0, stream>>>(a);
  }
  // ---- aspect BiLSTM recurrence (32 steps) ----
  for (int t = 0; t < LA_; ++t) {
    LstmArgs a;
    a.G0 = Ga[0]; a.G1 = Ga[1]; a.W0 = whhb[2]; a.W1 = whhb[3];
    a.hin0 = Hat(2, t & 1);  a.hin1 = Hat(3, t & 1);
    a.hout0 = Hat(2, (t + 1) & 1); a.hout1 = Hat(3, (t + 1) & 1);
    a.c0 = Cat(2); a.c1 = Cat(3);
    a.out = asp_out; a.lens = lens + B_; a.leftlen = lens + 2 * B_; a.asplen = lens + B_;
    a.T = LA_; a.t = t; a.use_locw = 0;
    k_lstm_step<<<dim3(H_ / 32, 1, 2), 256, 0, stream>>>(a);
  }

  // ---- pools & attention ----
  k_pool<<<B_, 256, 0, stream>>>(ctx_out, lens, ctx_pool, LC_);
  k_pool<<<B_, 256, 0, stream>>>(asp_out, lens + B_, asp_pool, LA_);
  k_matvec_u<<<dim3(H2_ / 256, B_), 256, 0, stream>>>(w_a2c, asp_pool, u_a2c);
  k_matvec_u<<<dim3(H2_ / 256, B_), 256, 0, stream>>>(w_c2a, ctx_pool, u_c2a);
  k_proj_w<<<B_, 256, 0, stream>>>(ctx_out, w_u, 0, cw1, LC_);            // w1
  k_proj_w<<<B_, 256, 0, stream>>>(asp_out, w_u + H2_, 0, aw2, LA_);      // w2
  k_scale_w3<<<(B_ * LA_ * H2_) / 256, 256, 0, stream>>>(asp_out, w_u + 2 * H2_, aspw3);
  k_align_gemm<<<dim3(LA_ / 16, LC_ / 128, B_), 256, 0, stream>>>(ctx_out, aspw3, cw1, aw2, align);
  k_rowmax<<<B_, 256, 0, stream>>>(align, rmax);
  k_colmax<<<B_, 256, 0, stream>>>(align, cmax);
  k_proj_w<<<B_, 256, 0, stream>>>(ctx_out, u_a2c, H2_, sc_a2c, LC_);
  k_proj_w<<<B_, 256, 0, stream>>>(asp_out, u_c2a, H2_, sc_c2a, LA_);
  k_softmax_wsum<<<B_, 256, 0, stream>>>(rmax, ctx_out, f_a2c, LC_);
  k_softmax_wsum<<<B_, 256, 0, stream>>>(cmax, asp_out, f_c2a, LA_);
  k_softmax_wsum<<<B_, 256, 0, stream>>>(sc_a2c, ctx_out, c_a2c, LC_);
  k_softmax_wsum<<<B_, 256, 0, stream>>>(sc_c2a, asp_out, c_c2a, LA_);
  k_dense<<<B_, 256, 0, stream>>>(c_a2c, f_a2c, f_c2a, c_c2a, dw, db, (float*)d_out);
}
